// LeanSelfAttention_43499428774270
// MI455X (gfx1250) — compile-verified, matrix operand order checked by reading, not run on hardware
//
#include <hip/hip_runtime.h>

// ---------------------------------------------------------------------------
// LeanSelfAttention for MI455X (gfx1250, wave32, WMMA bf16 + TDM async tiles)
// B=2, S=2048, HIDDEN=1024, HEADS=16, HEAD_DIM=64
// ---------------------------------------------------------------------------

#define B_ 2
#define S_ 2048
#define HID_ 1024
#define HEADS_ 16
#define HD_ 64
#define M_ (B_ * S_)          // 4096 tokens
#define NQKV_ (3 * HID_)      // 3072

typedef __attribute__((ext_vector_type(16))) __bf16 bf16x16;
typedef __attribute__((ext_vector_type(8)))  __bf16 bf16x8;
typedef __attribute__((ext_vector_type(8)))  float  f32x8;
typedef __attribute__((ext_vector_type(4)))  unsigned int u32x4;
typedef __attribute__((ext_vector_type(8)))  int i32x8;
typedef __attribute__((ext_vector_type(4)))  int i32x4;

#if defined(__has_builtin)
#if __has_builtin(__builtin_amdgcn_tensor_load_to_lds) && \
    __has_builtin(__builtin_amdgcn_s_wait_tensorcnt)
#define HAVE_TDM 1
#endif
#endif
#ifndef HAVE_TDM
#define HAVE_TDM 0
#endif

__device__ __forceinline__ bf16x16 combine8(bf16x8 lo, bf16x8 hi) {
  bf16x16 r;
#pragma unroll
  for (int t = 0; t < 8; ++t) { r[t] = lo[t]; r[t + 8] = hi[t]; }
  return r;
}

__device__ __forceinline__ f32x8 wmma_bf16(bf16x16 a, bf16x16 b, f32x8 c) {
  return __builtin_amdgcn_wmma_f32_16x16x32_bf16(false, a, false, b,
                                                 (short)0, c, false, false);
}

#if HAVE_TDM
// ---------------------------------------------------------------------------
// Tensor Data Mover: async 2D bf16 tile load (global, row-major) -> LDS with
// row padding.  D# bitfields per CDNA5 ISA.
//   pad_interval code: interval = 2<<code DWORDs; pad_amount code: (code+1) DW
// ---------------------------------------------------------------------------
__device__ __forceinline__ void tdm_load_tile_bf16(
    const __bf16* gsrc, const __bf16* lds_dst,
    unsigned tile_d0, unsigned tile_d1,       // tile size, elements
    unsigned long long stride0_elems,         // tensor dim0 stride, elements
    unsigned pad_interval_code, unsigned pad_amount_code) {
  unsigned long long ga = (unsigned long long)(size_t)gsrc;
  unsigned lds_addr = (unsigned)(size_t)lds_dst;  // LDS offset = addr[31:0]
  u32x4 g0;
  g0[0] = 1u;                                      // count=1, user descriptor
  g0[1] = lds_addr;                                // [63:32] lds_addr
  g0[2] = (unsigned)(ga & 0xffffffffu);            // [95:64] global_addr lo
  g0[3] = (unsigned)((ga >> 32) & 0x01ffffffu)     // [120:96] global_addr hi
        | (2u << 30);                              // [127:126] type=2 (image)
  i32x8 g1;
  g1[0] = (int)((1u << 16)                         // data_size=1 -> 2 bytes
              | (1u << 20)                         // pad_enable
              | (pad_interval_code << 22)
              | (pad_amount_code << 25));
  // tensor_dim0 [79:48] = tile_d0, tensor_dim1 [111:80] = tile_d1 (in-bounds)
  g1[1] = (int)((tile_d0 & 0xffffu) << 16);
  g1[2] = (int)(((tile_d0 >> 16) & 0xffffu) | ((tile_d1 & 0xffffu) << 16));
  g1[3] = (int)(((tile_d1 >> 16) & 0xffffu) | ((tile_d0 & 0xffffu) << 16)); // tile_dim0 [127:112]
  g1[4] = (int)(tile_d1 & 0xffffu);                // tile_dim1 [143:128], tile_dim2=0
  g1[5] = (int)(stride0_elems & 0xffffffffu);      // stride0 [191:160]
  g1[6] = (int)((stride0_elems >> 32) & 0xffffu);  // stride0 hi [207:192]
  g1[7] = 0;
  i32x4 z4 = {0, 0, 0, 0};
#if __has_include(<hip/amd_detail/amd_gfx1250_TDM.h>)
  i32x8 z8 = {0, 0, 0, 0, 0, 0, 0, 0};
  __builtin_amdgcn_tensor_load_to_lds(g0, g1, z4, z4, z8, 0);
#else
  __builtin_amdgcn_tensor_load_to_lds(g0, g1, z4, z4, 0);
#endif
}
#define TDM_FENCE() asm volatile("" ::: "memory")
#endif  // HAVE_TDM

// ---------------------------------------------------------------------------
// Kernel 1: LayerNorm (fp32 stats) -> bf16 activations
// ---------------------------------------------------------------------------
__global__ __launch_bounds__(256) void ln_bf16_kernel(
    const float* __restrict__ x, const float* __restrict__ g,
    const float* __restrict__ bb, __bf16* __restrict__ y) {
  const int tok = blockIdx.x;
  const float* xp = x + (size_t)tok * HID_;
  const int tid = threadIdx.x;
  float4 f4 = ((const float4*)xp)[tid];
  float vv[4] = {f4.x, f4.y, f4.z, f4.w};

  float s = vv[0] + vv[1] + vv[2] + vv[3];
#pragma unroll
  for (int off = 16; off; off >>= 1) s += __shfl_xor(s, off);
  __shared__ float red1[8], red2[8];
  const int wv = tid >> 5, ln = tid & 31;
  if (ln == 0) red1[wv] = s;
  __syncthreads();
  float tot = 0.f;
#pragma unroll
  for (int i = 0; i < 8; ++i) tot += red1[i];
  const float mean = tot * (1.0f / HID_);

  float sq = 0.f;
#pragma unroll
  for (int e = 0; e < 4; ++e) { float d = vv[e] - mean; sq += d * d; }
#pragma unroll
  for (int off = 16; off; off >>= 1) sq += __shfl_xor(sq, off);
  if (ln == 0) red2[wv] = sq;
  __syncthreads();
  float tot2 = 0.f;
#pragma unroll
  for (int i = 0; i < 8; ++i) tot2 += red2[i];
  const float rstd = rsqrtf(tot2 * (1.0f / HID_) + 1e-12f);

  const int i0 = tid * 4;
#pragma unroll
  for (int e = 0; e < 4; ++e) {
    float o = (vv[e] - mean) * rstd * g[i0 + e] + bb[i0 + e];
    y[(size_t)tok * HID_ + i0 + e] = (__bf16)o;
  }
}

// ---------------------------------------------------------------------------
// Kernel 2: fp32 [K][N] -> bf16 transposed [N][K]
// ---------------------------------------------------------------------------
__global__ __launch_bounds__(256) void transpose_bf16_kernel(
    const float* __restrict__ src, __bf16* __restrict__ dst, int K, int N) {
  __shared__ float tile[32][33];
  const int n0 = blockIdx.x * 32, k0 = blockIdx.y * 32;
  const int tx = threadIdx.x & 31, ty = threadIdx.x >> 5;
#pragma unroll
  for (int i = 0; i < 32; i += 8)
    tile[ty + i][tx] = src[(size_t)(k0 + ty + i) * N + n0 + tx];
  __syncthreads();
#pragma unroll
  for (int i = 0; i < 32; i += 8)
    dst[(size_t)(n0 + ty + i) * K + k0 + tx] = (__bf16)tile[tx][ty + i];
}

// ---------------------------------------------------------------------------
// Shared GEMM machinery: C[64x64]/block, 4 waves x (32x32), k-step 32.
// LDS rows padded to 40 bf16 (16 DW data + 4 DW pad -> TDM codes 3/3).
// ---------------------------------------------------------------------------
#define LDT 40

struct WaveAcc { f32x8 c[2][2]; };

__device__ __forceinline__ void gemm_mainloop(
    const __bf16* __restrict__ A, const __bf16* __restrict__ Bt, int K,
    int m0, int n0, __bf16* As, __bf16* Bs, WaveAcc& acc) {
  const int tid  = threadIdx.x;
  const int wave = tid >> 5, lane = tid & 31;
  const int wm = (wave >> 1) * 32, wn = (wave & 1) * 32;
  const int half = lane >> 4, lr = lane & 15;
  const int koff = half * 8;

  for (int k0 = 0; k0 < K; k0 += 32) {
#if HAVE_TDM
    if (wave == 0) {
      // 64 rows x 32 bf16 each, row padded by 4 DWORDs in LDS
      tdm_load_tile_bf16(A  + (size_t)m0 * K + k0, As, 32, 64,
                         (unsigned long long)K, 3u, 3u);
      tdm_load_tile_bf16(Bt + (size_t)n0 * K + k0, Bs, 32, 64,
                         (unsigned long long)K, 3u, 3u);
      __builtin_amdgcn_s_wait_tensorcnt(0);
    }
    TDM_FENCE();
#else
#pragma unroll
    for (int i = 0; i < 2; ++i) {          // 256 chunks of 8 bf16 per operand
      int chunk = tid + i * 128;
      int r = chunk >> 2, cc = (chunk & 3) * 8;
      *(bf16x8*)&As[r * LDT + cc] = *(const bf16x8*)&A[(size_t)(m0 + r) * K + k0 + cc];
      *(bf16x8*)&Bs[r * LDT + cc] = *(const bf16x8*)&Bt[(size_t)(n0 + r) * K + k0 + cc];
    }
#endif
    __syncthreads();

    bf16x16 a[2], b[2];
#pragma unroll
    for (int i = 0; i < 2; ++i) {
      int ra = wm + i * 16 + lr;
      a[i] = combine8(*(bf16x8*)&As[ra * LDT + koff],
                      *(bf16x8*)&As[ra * LDT + koff + 16]);
      int rb = wn + i * 16 + lr;
      b[i] = combine8(*(bf16x8*)&Bs[rb * LDT + koff],
                      *(bf16x8*)&Bs[rb * LDT + koff + 16]);
    }
#pragma unroll
    for (int i = 0; i < 2; ++i)
#pragma unroll
      for (int j = 0; j < 2; ++j)
        acc.c[i][j] = wmma_bf16(a[i], b[j], acc.c[i][j]);
    __syncthreads();
  }
}

// ---------------------------------------------------------------------------
// Kernel 3: QKV projection -> per-head bf16 Q/K/V [B][H][S][64]; K * 1/8.
// ---------------------------------------------------------------------------
__global__ __launch_bounds__(128) void gemm_qkv_kernel(
    const __bf16* __restrict__ xln, const __bf16* __restrict__ wT,
    const float* __restrict__ bias, __bf16* __restrict__ Q,
    __bf16* __restrict__ Kd, __bf16* __restrict__ V) {
  __shared__ __bf16 As[64 * LDT];
  __shared__ __bf16 Bs[64 * LDT];
  const int m0 = blockIdx.x * 64, n0 = blockIdx.y * 64;
  WaveAcc acc = {};
  gemm_mainloop(xln, wT, HID_, m0, n0, As, Bs, acc);

  const int tid = threadIdx.x, wave = tid >> 5, lane = tid & 31;
  const int wm = (wave >> 1) * 32, wn = (wave & 1) * 32;
  const int half = lane >> 4, lr = lane & 15;
#pragma unroll
  for (int i = 0; i < 2; ++i)
#pragma unroll
    for (int j = 0; j < 2; ++j) {
      const int gn = n0 + wn + j * 16 + lr;           // 0..3071
      const int sel = gn >> 10;                        // 0=q 1=k 2=v
      const int n1 = gn & (HID_ - 1);
      const int h = n1 >> 6, d = n1 & 63;
      __bf16* dst = (sel == 0) ? Q : (sel == 1) ? Kd : V;
      const float scale = (sel == 1) ? 0.125f : 1.0f;  // 1/sqrt(64)
      const float bv = bias[gn];
#pragma unroll
      for (int r = 0; r < 8; ++r) {
        const int gm = m0 + wm + i * 16 + half * 8 + r; // token
        const int bb = gm >> 11, ss = gm & (S_ - 1);
        float val = (acc.c[i][j][r] + bv) * scale;
        dst[(((size_t)(bb * HEADS_ + h) * S_ + ss) << 6) + d] = (__bf16)val;
      }
    }
}

// ---------------------------------------------------------------------------
// Kernel 4: flash attention. Block = (64 q-rows, one (b,h)); K tile via TDM,
// V^T tile manual (transpose), online softmax, P restaged via per-wave LDS.
// ---------------------------------------------------------------------------
#define KLDT 72   // 64 + 8 pad (32 DW data + 4 DW pad -> TDM codes 4/3)
#define PLDT 40   // 32 + 8 pad

__global__ __launch_bounds__(128) void attn_kernel(
    const __bf16* __restrict__ Q, const __bf16* __restrict__ Kb,
    const __bf16* __restrict__ Vb, const float* __restrict__ mask,
    __bf16* __restrict__ attn_out) {
  __shared__ __bf16 Ks[32 * KLDT];
  __shared__ __bf16 Vts[64 * PLDT];
  __shared__ __bf16 Ps[4][16 * PLDT];

  const int tid = threadIdx.x, wave = tid >> 5, lane = tid & 31;
  const int half = lane >> 4, lr = lane & 15;
  const int koff = half * 8;
  const int q0 = blockIdx.x * 64;
  const int bh = blockIdx.y;                 // b*16 + h
  const int b = bh >> 4, h = bh & 15;

  const __bf16* Qp = Q  + (size_t)bh * S_ * HD_;
  const __bf16* Kp = Kb + (size_t)bh * S_ * HD_;
  const __bf16* Vp = Vb + (size_t)bh * S_ * HD_;
  const float*  Mp = mask + (size_t)b * S_ * S_;

  // Q fragments for this wave's 16 rows (K-dim 64 -> two 16x32 A operands)
  const int qrow = q0 + wave * 16 + lr;
  bf16x16 aq[2];
#pragma unroll
  for (int kk = 0; kk < 2; ++kk)
    aq[kk] = combine8(*(const bf16x8*)&Qp[(size_t)qrow * HD_ + kk * 32 + koff],
                      *(const bf16x8*)&Qp[(size_t)qrow * HD_ + kk * 32 + koff + 16]);

  f32x8 o[4] = {};                  // output accum: 4 d-subtiles of 16
  float mrow[8], lrow[8];
#pragma unroll
  for (int r = 0; r < 8; ++r) { mrow[r] = -1e30f; lrow[r] = 0.f; }

  for (int kb = 0; kb < S_; kb += 32) {
#if HAVE_TDM
    if (wave == 0)      // K tile: 32 keys x 64 bf16, padded rows in LDS
      tdm_load_tile_bf16(Kp + (size_t)kb * HD_, Ks, 64, 32,
                         (unsigned long long)HD_, 4u, 3u);
#else
#pragma unroll
    for (int i = 0; i < 2; ++i) {
      int chunk = tid + i * 128;
      int r = chunk >> 3, cc = (chunk & 7) * 8;
      *(bf16x8*)&Ks[r * KLDT + cc] =
          *(const bf16x8*)&Kp[(size_t)(kb + r) * HD_ + cc];
    }
#endif
    // cooperative V^T tile: Vts[d][key]
#pragma unroll
    for (int i = 0; i < 2; ++i) {
      int chunk = tid + i * 128;
      int r = chunk >> 3, cc = (chunk & 7) * 8;
      bf16x8 t = *(const bf16x8*)&Vp[(size_t)(kb + r) * HD_ + cc];
#pragma unroll
      for (int e = 0; e < 8; ++e) Vts[(cc + e) * PLDT + r] = t[e];
    }
    if (kb + 32 < S_) __builtin_prefetch(&Kp[(size_t)(kb + 32) * HD_], 0, 3);
#if HAVE_TDM
    if (wave == 0) __builtin_amdgcn_s_wait_tensorcnt(0);
    TDM_FENCE();
#endif
    __syncthreads();

    // scores: two 16x16 tiles (keys 0-15 / 16-31), each K-dim 64 -> 2 WMMAs
    f32x8 s0 = {}, s1 = {};
#pragma unroll
    for (int kk = 0; kk < 2; ++kk) {
      int row0 = lr, row1 = 16 + lr;
      bf16x16 b0 = combine8(*(bf16x8*)&Ks[row0 * KLDT + kk * 32 + koff],
                            *(bf16x8*)&Ks[row0 * KLDT + kk * 32 + koff + 16]);
      bf16x16 b1 = combine8(*(bf16x8*)&Ks[row1 * KLDT + kk * 32 + koff],
                            *(bf16x8*)&Ks[row1 * KLDT + kk * 32 + koff + 16]);
      s0 = wmma_bf16(aq[kk], b0, s0);
      s1 = wmma_bf16(aq[kk], b1, s1);
    }

    // online softmax update (row = r + half*8 for each VGPR slot r)
    float p0[8], p1[8];
#pragma unroll
    for (int r = 0; r < 8; ++r) {
      const int qm = q0 + wave * 16 + half * 8 + r;
      float x0 = s0[r] + Mp[(size_t)qm * S_ + kb + lr];
      float x1 = s1[r] + Mp[(size_t)qm * S_ + kb + 16 + lr];
      float t = fmaxf(x0, x1);
#pragma unroll
      for (int off = 8; off; off >>= 1) t = fmaxf(t, __shfl_xor(t, off));
      const float mnew = fmaxf(mrow[r], t);
      const float alpha = __expf(mrow[r] - mnew);
      p0[r] = __expf(x0 - mnew);
      p1[r] = __expf(x1 - mnew);
      float rs = p0[r] + p1[r];
#pragma unroll
      for (int off = 8; off; off >>= 1) rs += __shfl_xor(rs, off);
      lrow[r] = lrow[r] * alpha + rs;
      mrow[r] = mnew;
#pragma unroll
      for (int j = 0; j < 4; ++j) o[j][r] *= alpha;
      const int m = half * 8 + r;
      Ps[wave][m * PLDT + lr]      = (__bf16)p0[r];
      Ps[wave][m * PLDT + 16 + lr] = (__bf16)p1[r];
    }

    // P (16x32) as A; V^T rows as B: O += P x V
    bf16x16 pa = combine8(*(bf16x8*)&Ps[wave][lr * PLDT + koff],
                          *(bf16x8*)&Ps[wave][lr * PLDT + koff + 16]);
#pragma unroll
    for (int j = 0; j < 4; ++j) {
      const int dn = j * 16 + lr;
      bf16x16 bv = combine8(*(bf16x8*)&Vts[dn * PLDT + koff],
                            *(bf16x8*)&Vts[dn * PLDT + koff + 16]);
      o[j] = wmma_bf16(pa, bv, o[j]);
    }
    __syncthreads();
  }

  // finalize: O /= l, write bf16 attn [token][h*64 + d]
#pragma unroll
  for (int r = 0; r < 8; ++r) {
    const int ss = q0 + wave * 16 + half * 8 + r;
    const float inv = 1.0f / lrow[r];
    const size_t base = (size_t)(b * S_ + ss) * HID_ + h * HD_;
#pragma unroll
    for (int j = 0; j < 4; ++j)
      attn_out[base + j * 16 + lr] = (__bf16)(o[j][r] * inv);
  }
}

// ---------------------------------------------------------------------------
// Kernel 5: output projection + bias + residual -> fp32 out
// ---------------------------------------------------------------------------
__global__ __launch_bounds__(128) void gemm_out_kernel(
    const __bf16* __restrict__ attn, const __bf16* __restrict__ wT,
    const float* __restrict__ bias, const float* __restrict__ resid,
    float* __restrict__ out) {
  __shared__ __bf16 As[64 * LDT];
  __shared__ __bf16 Bs[64 * LDT];
  const int m0 = blockIdx.x * 64, n0 = blockIdx.y * 64;
  WaveAcc acc = {};
  gemm_mainloop(attn, wT, HID_, m0, n0, As, Bs, acc);

  const int tid = threadIdx.x, wave = tid >> 5, lane = tid & 31;
  const int wm = (wave >> 1) * 32, wn = (wave & 1) * 32;
  const int half = lane >> 4, lr = lane & 15;
#pragma unroll
  for (int i = 0; i < 2; ++i)
#pragma unroll
    for (int j = 0; j < 2; ++j) {
      const int gn = n0 + wn + j * 16 + lr;
      const float bv = bias[gn];
#pragma unroll
      for (int r = 0; r < 8; ++r) {
        const int gm = m0 + wm + i * 16 + half * 8 + r;
        out[(size_t)gm * HID_ + gn] =
            acc.c[i][j][r] + bv + resid[(size_t)gm * HID_ + gn];
      }
    }
}

// ---------------------------------------------------------------------------
// Host launcher
// ---------------------------------------------------------------------------
extern "C" void kernel_launch(void* const* d_in, const int* in_sizes, int n_in,
                              void* d_out, int out_size, void* d_ws,
                              size_t ws_size, hipStream_t stream) {
  const float* hidden = (const float*)d_in[0];
  const float* mask   = (const float*)d_in[1];
  const float* w_qkv  = (const float*)d_in[2];
  const float* b_qkv  = (const float*)d_in[3];
  const float* w_out  = (const float*)d_in[4];
  const float* b_out  = (const float*)d_in[5];
  const float* ln_g   = (const float*)d_in[6];
  const float* ln_b   = (const float*)d_in[7];
  float* out = (float*)d_out;

  char* p = (char*)d_ws;
  __bf16* xln   = (__bf16*)p; p += (size_t)M_ * HID_ * 2;         // 8 MB
  __bf16* wqkvT = (__bf16*)p; p += (size_t)NQKV_ * HID_ * 2;      // 6 MB
  __bf16* woutT = (__bf16*)p; p += (size_t)HID_ * HID_ * 2;       // 2 MB
  __bf16* Qw    = (__bf16*)p; p += (size_t)B_ * HEADS_ * S_ * HD_ * 2;
  __bf16* Kw    = (__bf16*)p; p += (size_t)B_ * HEADS_ * S_ * HD_ * 2;
  __bf16* Vw    = (__bf16*)p; p += (size_t)B_ * HEADS_ * S_ * HD_ * 2;
  __bf16* attnw = (__bf16*)p; p += (size_t)M_ * HID_ * 2;         // 8 MB

  ln_bf16_kernel<<<M_, 256, 0, stream>>>(hidden, ln_g, ln_b, xln);
  transpose_bf16_kernel<<<dim3(NQKV_ / 32, HID_ / 32), 256, 0, stream>>>(
      w_qkv, wqkvT, HID_, NQKV_);
  transpose_bf16_kernel<<<dim3(HID_ / 32, HID_ / 32), 256, 0, stream>>>(
      w_out, woutT, HID_, HID_);
  gemm_qkv_kernel<<<dim3(M_ / 64, NQKV_ / 64), 128, 0, stream>>>(
      xln, wqkvT, b_qkv, Qw, Kw, Vw);
  attn_kernel<<<dim3(S_ / 64, B_ * HEADS_), 128, 0, stream>>>(
      Qw, Kw, Vw, mask, attnw);
  gemm_out_kernel<<<dim3(M_ / 64, HID_ / 64), 128, 0, stream>>>(
      attnw, woutT, b_out, hidden, out);
}
